// VolumeRenderer_52106543235858
// MI455X (gfx1250) — matrix operand order repaired and checked by reading, not compiled, not run on hardware
//
#include <hip/hip_runtime.h>
#include <math.h>

typedef __attribute__((ext_vector_type(16))) _Float16 v16h;
typedef __attribute__((ext_vector_type(8)))  float    v8f;

#define GRID_N   128
#define CHANNELS 28
#define N_STEPS  128
#define RAYS_PER_BLOCK 8

__device__ __forceinline__ float bcast15(float x) {
    return __int_as_float(__builtin_amdgcn_readlane(__float_as_int(x), 15));
}
__device__ __forceinline__ float bpermf(int byteaddr, float x) {
    return __int_as_float(__builtin_amdgcn_ds_bpermute(byteaddr, __float_as_int(x)));
}

__global__ __launch_bounds__(256) void
volume_render_wmma(const float* __restrict__ origins,
                   const float* __restrict__ dirs,
                   const float* __restrict__ viewdirs,
                   const float* __restrict__ grid,
                   float* __restrict__ out, int B)
{
    const int lane = threadIdx.x & 31;
    const int wave = threadIdx.x >> 5;
    const int ray  = blockIdx.x * RAYS_PER_BLOCK + wave;
    if (ray >= B) return;                      // wave-uniform exit

    // ---- per-ray setup (uniform across the wave) --------------------------
    float ox = origins[3*ray+0], oy = origins[3*ray+1], oz = origins[3*ray+2];
    float dx = dirs[3*ray+0],    dy = dirs[3*ray+1],    dz = dirs[3*ray+2];
    float vx = viewdirs[3*ray+0],vy = viewdirs[3*ray+1],vz = viewdirs[3*ray+2];

    float dn = 1.0f / sqrtf(dx*dx + dy*dy + dz*dz);
    dx *= dn; dy *= dn; dz *= dn;
    float vn = 1.0f / sqrtf(vx*vx + vy*vy + vz*vz);
    vx *= vn; vy *= vn; vz *= vn;

    // SH2 basis
    const float SH_C0 = 0.28209479177387814f;
    const float SH_C1 = 0.4886025119029199f;
    const float C20 = 1.0925484305920792f, C21 = -1.0925484305920792f;
    const float C22 = 0.31539156525252005f, C23 = -1.0925484305920792f;
    const float C24 = 0.5462742152960396f;
    float sh0 = SH_C0;
    float sh1 = -SH_C1 * vy;
    float sh2 =  SH_C1 * vz;
    float sh3 = -SH_C1 * vx;
    float sh4 =  C20 * vx * vy;
    float sh5 =  C21 * vy * vz;
    float sh6 =  C22 * (2.0f*vz*vz - vx*vx - vy*vy);
    float sh7 =  C23 * vx * vz;
    float sh8 =  C24 * (vx*vx - vy*vy);

    // ray-box intersection (matches reference exactly)
    float ivx = 1.0f/(dx + 1e-9f), ivy = 1.0f/(dy + 1e-9f), ivz = 1.0f/(dz + 1e-9f);
    float t1x = (0.0f-ox)*ivx, t2x = (1.0f-ox)*ivx;
    float t1y = (0.0f-oy)*ivy, t2y = (1.0f-oy)*ivy;
    float t1z = (0.0f-oz)*ivz, t2z = (1.0f-oz)*ivz;
    float tmin = fmaxf(fmaxf(fminf(t1x,t2x), fminf(t1y,t2y)), fminf(t1z,t2z));
    tmin = fmaxf(tmin, 0.0f);
    float tmax = fminf(fminf(fmaxf(t1x,t2x), fmaxf(t1y,t2y)), fmaxf(t1z,t2z));
    tmax = fmaxf(tmax, tmin);
    float dt = (tmax - tmin) * (1.0f / (float)N_STEPS);

    // ---- build per-ray constant A matrix (16x32 f16) ----------------------
    // row m<3 : block-diagonal SH basis   A[m][9m+k] = sh[k]
    // row m=3 : A[3][27] = 1   (passes sigma through the WMMA)
    // A layout (16-bit A, 16x32): lanes 0-15 hold K=0..15 (elem 0-15);
    // lanes 16-31 hold K=16..31 (matches B's contiguous channel order).
    const bool hi = (lane >= 16);
    const int  m  = lane & 15;
    v16h Amat;
#pragma unroll
    for (int e = 0; e < 16; ++e) {
        int k = e + (hi ? 16 : 0);
        float v = 0.0f;
        if (m < 3) {
            int kk = k - 9*m;
            if (kk >= 0 && kk < 9) {
                v = (kk==0)?sh0:(kk==1)?sh1:(kk==2)?sh2:(kk==3)?sh3:
                    (kk==4)?sh4:(kk==5)?sh5:(kk==6)?sh6:(kk==7)?sh7:sh8;
            }
        } else if (m == 3 && k == 27) {
            v = 1.0f;
        }
        Amat[e] = (_Float16)v;
    }

    // ---- loop-invariant scan plumbing: bpermute byte-addrs + float masks --
    const int   a1 = ((lane - 1) & 31) << 2;
    const int   a2 = ((lane - 2) & 31) << 2;
    const int   a4 = ((lane - 4) & 31) << 2;
    const int   a8 = ((lane - 8) & 31) << 2;
    const float fm1 = (lane >= 1) ? 1.0f : 0.0f;
    const float fm2 = (lane >= 2) ? 1.0f : 0.0f;
    const float fm4 = (lane >= 4) ? 1.0f : 0.0f;
    const float fm8 = (lane >= 8) ? 1.0f : 0.0f;

    // ---- march: 8 rounds of 16 steps, one WMMA per round ------------------
    float light = 1.0f;
    float r_acc = 0.0f, g_acc = 0.0f, b_acc = 0.0f;
    const int scol = lane & 15;            // this lane's step-column
    const float Gf = (float)GRID_N;

#pragma unroll
    for (int round = 0; round < 8; ++round) {
        // voxel for step (round*16 + scol)
        float t  = fmaf((float)(round*16 + scol) + 0.5f, dt, tmin);
        float px = fmaf(t, dx, ox), py = fmaf(t, dy, oy), pz = fmaf(t, dz, oz);
        int ix = min(max((int)floorf(px*Gf), 0), GRID_N-1);
        int iy = min(max((int)floorf(py*Gf), 0), GRID_N-1);
        int iz = min(max((int)floorf(pz*Gf), 0), GRID_N-1);
        int rowoff = ((ix*GRID_N + iy)*GRID_N + iz) * CHANNELS;

        // lanes 0-15 read ch 0..15; lanes 16-31 read ch 16..27 (in-row, 16B
        // aligned). 4th load for hi lanes re-reads rp[2] (same cacheline).
        const float4* rp = (const float4*)(grid + rowoff + (hi ? 16 : 0));
        float4 q0 = rp[0], q1 = rp[1], q2 = rp[2];
        float4 q3 = rp[hi ? 2 : 3];

        // B layout (32x16, 16-bit): lane n elem e = channel e ; lane n+16
        // elem e = channel 16+e (28..31 zero, killed by zero columns of A)
        float f12 = hi ? 0.f : q3.x;  float f13 = hi ? 0.f : q3.y;
        float f14 = hi ? 0.f : q3.z;  float f15 = hi ? 0.f : q3.w;

        v16h Bmat;
        Bmat[0]=(_Float16)q0.x;  Bmat[1]=(_Float16)q0.y;
        Bmat[2]=(_Float16)q0.z;  Bmat[3]=(_Float16)q0.w;
        Bmat[4]=(_Float16)q1.x;  Bmat[5]=(_Float16)q1.y;
        Bmat[6]=(_Float16)q1.z;  Bmat[7]=(_Float16)q1.w;
        Bmat[8]=(_Float16)q2.x;  Bmat[9]=(_Float16)q2.y;
        Bmat[10]=(_Float16)q2.z; Bmat[11]=(_Float16)q2.w;
        Bmat[12]=(_Float16)f12;  Bmat[13]=(_Float16)f13;
        Bmat[14]=(_Float16)f14;  Bmat[15]=(_Float16)f15;

        v8f Cz = {};
        v8f D = __builtin_amdgcn_wmma_f32_16x16x32_f16(
                    false, Amat, false, Bmat, (short)0, Cz, false, false);

        // lane n (<16): D[0..3] = r,g,b,sigma of step n.
        // lanes >=16 see rows 8-11 of D (exactly zero): att=1 -> weight=0.
        float red = D[0], grn = D[1], blu = D[2];
        float sig = fmaxf(D[3], 0.0f);

        // INCLUSIVE prefix-sum of sigma over the 16 step-lanes.
        // 4 unconditional ds_bpermutes; masks folded into FMAs (no branches,
        // no per-round v_cmp).  prefix-prod(att) = exp(-dt * prefix-sum(sig))
        float Sx = sig;
        Sx = fmaf(bpermf(a1, Sx), fm1, Sx);
        Sx = fmaf(bpermf(a2, Sx), fm2, Sx);
        Sx = fmaf(bpermf(a4, Sx), fm4, Sx);
        Sx = fmaf(bpermf(a8, Sx), fm8, Sx);
        float Sexcl = Sx - sig;                // exclusive prefix

        float att = __expf(-dt * sig);
        float T   = __expf(-dt * Sexcl);       // transmittance within round
        float cr  = __builtin_amdgcn_rcpf(1.0f + __expf(-red));
        float cg  = __builtin_amdgcn_rcpf(1.0f + __expf(-grn));
        float cb  = __builtin_amdgcn_rcpf(1.0f + __expf(-blu));

        float w = light * T * (1.0f - att);
        r_acc += w * cr;  g_acc += w * cg;  b_acc += w * cb;

        // round total transmittance = exp(-dt * sum of all 16 sigmas)
        float Stot = bcast15(Sx);              // v_readlane (uniform)
        light *= __expf(-dt * Stot);
    }

    // ---- reduce rgb over the 16 step-lanes (hi lanes hold exact zeros) ----
#pragma unroll
    for (int ofs = 8; ofs >= 1; ofs >>= 1) {
        r_acc += __shfl_xor(r_acc, ofs, 32);
        g_acc += __shfl_xor(g_acc, ofs, 32);
        b_acc += __shfl_xor(b_acc, ofs, 32);
    }

    if (lane == 0) {
        out[3*ray+0] = r_acc + light;   // BG_BRIGHTNESS = 1.0
        out[3*ray+1] = g_acc + light;
        out[3*ray+2] = b_acc + light;
    }
}

extern "C" void kernel_launch(void* const* d_in, const int* in_sizes, int n_in,
                              void* d_out, int out_size, void* d_ws, size_t ws_size,
                              hipStream_t stream)
{
    const float* origins  = (const float*)d_in[0];
    const float* dirs     = (const float*)d_in[1];
    const float* viewdirs = (const float*)d_in[2];
    const float* grid     = (const float*)d_in[3];
    float* out = (float*)d_out;

    int B = in_sizes[0] / 3;                       // 32768
    int blocks = (B + RAYS_PER_BLOCK - 1) / RAYS_PER_BLOCK;
    volume_render_wmma<<<blocks, 256, 0, stream>>>(origins, dirs, viewdirs,
                                                   grid, out, B);
}
